// MultiHeadSelfAttention_2680059592936
// MI455X (gfx1250) — matrix operand order repaired and checked by reading, not compiled, not run on hardware
//
#include <hip/hip_runtime.h>

typedef __attribute__((ext_vector_type(2)))  float  v2f;
typedef __attribute__((ext_vector_type(4)))  float  v4f;
typedef __attribute__((ext_vector_type(8)))  float  v8f;
typedef __attribute__((ext_vector_type(8)))  __bf16 v8bf;
typedef __attribute__((ext_vector_type(16))) __bf16 v16bf;

#define N_ 4
#define S_ 4096
#define E_ 4096
#define H_ 32
#define D_ 128

// ---- CDNA5 async global->LDS copy (ASYNCcnt path), inline asm = portable ----
__device__ __forceinline__ void async_ld_b128(unsigned lds_off, const void* g) {
  asm volatile("global_load_async_to_lds_b128 %0, %1, off"
               :: "v"(lds_off), "v"(g) : "memory");
}
__device__ __forceinline__ void wait_async0() {
  asm volatile("s_wait_asynccnt 0x0" ::: "memory");
}

// ---------------- Stage 0: W_out f32 -> bf16 (L2-resident operand) ----------
__global__ __launch_bounds__(256) void cvt_w_kernel(const float* __restrict__ w,
                                                    __bf16* __restrict__ wb) {
  size_t i = ((size_t)blockIdx.x * 256 + threadIdx.x) * 4;
  v4f x = *(const v4f*)(w + i);
  wb[i + 0] = (__bf16)x.x;
  wb[i + 1] = (__bf16)x.y;
  wb[i + 2] = (__bf16)x.z;
  wb[i + 3] = (__bf16)x.w;
}

// ---------------- Stage 1: per-position attention over heads ----------------
// One wave per (n,s). S = QK^T/sqrt(128) via v_wmma_f32_16x16x4_f32 (full f32),
// in-register softmax with lane shuffles, P transposed through LDS into
// A-fragment layout, O = P*V via f32 WMMA, output written bf16 to workspace.
__global__ __launch_bounds__(128) void attn_softmax_kernel(
    const float* __restrict__ Vm, const float* __restrict__ Km,
    const float* __restrict__ Qm, const int* __restrict__ mask,
    __bf16* __restrict__ Xbf) {
  __shared__ float Plds[4][32 * 34];  // stride 34 floats: 8B-aligned, bank-clean
  const int lane = threadIdx.x & 31;
  const int wv   = threadIdx.x >> 5;
  const int pos  = blockIdx.x * 4 + wv;   // 0 .. N*S-1
  const int n    = pos >> 12;             // S = 4096
  const int ln   = lane & 15;
  const int g    = lane >> 4;

  const float* q = Qm + (size_t)pos * E_;
  const float* k = Km + (size_t)pos * E_;
  const float* v = Vm + (size_t)pos * E_;

  // ---- scores: 2x2 tiles of 16x16, K = 128 in chunks of 4 ----
  v8f s[2][2];
#pragma unroll
  for (int ti = 0; ti < 2; ++ti)
#pragma unroll
    for (int tj = 0; tj < 2; ++tj)
#pragma unroll
      for (int r = 0; r < 8; ++r) s[ti][tj][r] = 0.0f;

#pragma unroll 4
  for (int kc = 0; kc < 32; ++kc) {
    const int kb = kc * 4 + 2 * g;  // two consecutive k per lane (b64 loads)
    v2f a[2], b[2];
#pragma unroll
    for (int ti = 0; ti < 2; ++ti)
      a[ti] = *(const v2f*)(q + (ln + 16 * ti) * D_ + kb);
#pragma unroll
    for (int tj = 0; tj < 2; ++tj)
      b[tj] = *(const v2f*)(k + (ln + 16 * tj) * D_ + kb);
#pragma unroll
    for (int ti = 0; ti < 2; ++ti)
#pragma unroll
      for (int tj = 0; tj < 2; ++tj)
        s[ti][tj] = __builtin_amdgcn_wmma_f32_16x16x4_f32(
            false, a[ti], false, b[tj], (short)0, s[ti][tj], false, false);
  }

  // ---- scale + mask + row softmax (rows live across 16-lane groups) ----
  const float scale = 0.08838834764831845f;  // 1/sqrt(128)
  const int* mrow = mask + n * H_ * H_;
#pragma unroll
  for (int ti = 0; ti < 2; ++ti) {
#pragma unroll
    for (int tj = 0; tj < 2; ++tj)
#pragma unroll
      for (int r = 0; r < 8; ++r) {
        float x = s[ti][tj][r] * scale;
        const int m = r + 8 * g + 16 * ti;
        const int j = ln + 16 * tj;
        if (mrow[m * H_ + j] == 0) x = -1.0e20f;
        s[ti][tj][r] = x;
      }
    float inv[8];
#pragma unroll
    for (int r = 0; r < 8; ++r) {
      float t = fmaxf(s[ti][0][r], s[ti][1][r]);
#pragma unroll
      for (int off = 8; off >= 1; off >>= 1)
        t = fmaxf(t, __shfl_xor(t, off, 32));
      const float mx = t;
      float e0 = __expf(s[ti][0][r] - mx);
      float e1 = __expf(s[ti][1][r] - mx);
      s[ti][0][r] = e0;
      s[ti][1][r] = e1;
      float sm = e0 + e1;
#pragma unroll
      for (int off = 8; off >= 1; off >>= 1)
        sm += __shfl_xor(sm, off, 32);
      inv[r] = 1.0f / sm;
    }
#pragma unroll
    for (int tj = 0; tj < 2; ++tj)
#pragma unroll
      for (int r = 0; r < 8; ++r) s[ti][tj][r] *= inv[r];
  }

  // ---- transpose D-layout probs -> A-fragment layout via LDS ----
  float* P = Plds[wv];
#pragma unroll
  for (int ti = 0; ti < 2; ++ti)
#pragma unroll
    for (int tj = 0; tj < 2; ++tj)
#pragma unroll
      for (int r = 0; r < 8; ++r)
        P[(r + 8 * g + 16 * ti) * 34 + ln + 16 * tj] = s[ti][tj][r];
  __syncthreads();

  v2f pa[8][2];  // A fragments for O = P*V, all k-chunks, both row tiles
#pragma unroll
  for (int jc = 0; jc < 8; ++jc) {
    const int jb = jc * 4 + 2 * g;
#pragma unroll
    for (int ti = 0; ti < 2; ++ti) {
      pa[jc][ti].x = P[(ln + 16 * ti) * 34 + jb];
      pa[jc][ti].y = P[(ln + 16 * ti) * 34 + jb + 1];
    }
  }

  // ---- O = P * V, 8 column tiles of 16; store bf16 for the projection ----
  __bf16* xp = Xbf + (size_t)pos * E_;
#pragma unroll 2
  for (int dt = 0; dt < 8; ++dt) {
    const int d = ln + 16 * dt;
    v8f o[2];
#pragma unroll
    for (int ti = 0; ti < 2; ++ti)
#pragma unroll
      for (int r = 0; r < 8; ++r) o[ti][r] = 0.0f;
#pragma unroll
    for (int jc = 0; jc < 8; ++jc) {
      const int jb = jc * 4 + 2 * g;
      v2f bv;
      bv.x = v[jb * D_ + d];
      bv.y = v[(jb + 1) * D_ + d];
#pragma unroll
      for (int ti = 0; ti < 2; ++ti)
        o[ti] = __builtin_amdgcn_wmma_f32_16x16x4_f32(
            false, pa[jc][ti], false, bv, (short)0, o[ti], false, false);
    }
#pragma unroll
    for (int ti = 0; ti < 2; ++ti)
#pragma unroll
      for (int r = 0; r < 8; ++r)
        xp[(r + 8 * g + 16 * ti) * D_ + d] = (__bf16)o[ti][r];
  }
}

// ---------------- Stage 2: Y = X * W^T + b, bf16 WMMA, f32 accumulate -------
// 8 waves per block as 2x4 wave-tiles -> 128x256 block tile. The shared
// A (128x32) / B (256x32) k-slices are staged into LDS with CDNA5 async
// global->LDS b128 copies, double buffered so the DMA overlaps the WMMAs.
// Cuts L2 traffic ~5x vs per-wave global fragment loads.
#define BM_ 128
#define BE_ 256
#define LST_ 40  // bf16 elems per LDS row (80 B: 16B aligned, conflict-free)

__global__ __launch_bounds__(256) void out_proj_kernel(
    const __bf16* __restrict__ X, const __bf16* __restrict__ Wb,
    const float* __restrict__ bias, float* __restrict__ Y) {
  __shared__ __bf16 Alds[2][BM_ * LST_];
  __shared__ __bf16 Blds[2][BE_ * LST_];

  const int tid  = threadIdx.x;
  const int lane = tid & 31;
  const int ln   = lane & 15;
  const int g    = lane >> 4;
  const int ww   = tid >> 5;  // 0..7
  const int wml  = ww >> 2;   // 0..1  (M wave tile)
  const int wel  = ww & 3;    // 0..3  (E wave tile)

  const int be = blockIdx.x & 15;  // 16 E-blocks of 256
  const int bm = blockIdx.x >> 4;  // 128 M-blocks of 128
  const size_t m0b = (size_t)bm * BM_;
  const int e0b = be * BE_;

  const unsigned aBase[2] = {(unsigned)(size_t)&Alds[0][0],
                             (unsigned)(size_t)&Alds[1][0]};
  const unsigned bBase[2] = {(unsigned)(size_t)&Blds[0][0],
                             (unsigned)(size_t)&Blds[1][0]};

  // Async stage of one 32-wide k-slice: A = 512 16B chunks (2/thread),
  // B = 1024 16B chunks (4/thread).
  auto issue = [&](int kk, int buf) {
#pragma unroll
    for (int t = 0; t < 2; ++t) {
      const int c = tid * 2 + t;
      const int row = c >> 2, ch = c & 3;
      async_ld_b128(aBase[buf] + (unsigned)(row * (LST_ * 2) + ch * 16),
                    X + (m0b + row) * (size_t)E_ + kk + ch * 8);
    }
#pragma unroll
    for (int t = 0; t < 4; ++t) {
      const int c = tid * 4 + t;
      const int row = c >> 2, ch = c & 3;
      async_ld_b128(bBase[buf] + (unsigned)(row * (LST_ * 2) + ch * 16),
                    Wb + (size_t)(e0b + row) * E_ + kk + ch * 8);
    }
  };

  v8f acc[4][4];
#pragma unroll
  for (int mi = 0; mi < 4; ++mi)
#pragma unroll
    for (int ei = 0; ei < 4; ++ei)
#pragma unroll
      for (int r = 0; r < 8; ++r) acc[mi][ei][r] = 0.0f;

  issue(0, 0);
  wait_async0();
  __syncthreads();

  for (int kk = 0; kk < E_; kk += 32) {
    const int buf = (kk >> 5) & 1;
    if (kk + 32 < E_) issue(kk + 32, buf ^ 1);  // DMA overlaps WMMA below

    v16bf a[4], b[4];
#pragma unroll
    for (int mi = 0; mi < 4; ++mi) {  // A 16x32 bf16: two 8-elem runs per lane
      const __bf16* ap = &Alds[buf][(wml * 64 + mi * 16 + ln) * LST_];
      v8bf lo = *(const v8bf*)(ap + 8 * g);
      v8bf hi = *(const v8bf*)(ap + 16 + 8 * g);
      a[mi] = __builtin_shufflevector(lo, hi, 0, 1, 2, 3, 4, 5, 6, 7, 8, 9, 10,
                                      11, 12, 13, 14, 15);
    }
#pragma unroll
    for (int ei = 0; ei < 4; ++ei) {  // B 32x16 bf16: two 8-elem runs per lane
      const __bf16* bp = &Blds[buf][(wel * 64 + ei * 16 + ln) * LST_];
      v8bf lo = *(const v8bf*)(bp + 8 * g);
      v8bf hi = *(const v8bf*)(bp + 16 + 8 * g);
      b[ei] = __builtin_shufflevector(lo, hi, 0, 1, 2, 3, 4, 5, 6, 7, 8, 9, 10,
                                      11, 12, 13, 14, 15);
    }
#pragma unroll
    for (int mi = 0; mi < 4; ++mi)
#pragma unroll
      for (int ei = 0; ei < 4; ++ei)
        acc[mi][ei] = __builtin_amdgcn_wmma_f32_16x16x32_bf16(
            false, a[mi], false, b[ei], (short)0, acc[mi][ei], false, false);

    wait_async0();   // next k-slice landed
    __syncthreads(); // all waves done reading buf / writing buf^1
  }

  // epilogue: +bias, f32 store
  const size_t m0 = m0b + (size_t)wml * 64;
  const int e0 = e0b + wel * 64;
#pragma unroll
  for (int ei = 0; ei < 4; ++ei) {
    const int e = e0 + 16 * ei + ln;
    const float bb = bias[e];
#pragma unroll
    for (int mi = 0; mi < 4; ++mi) {
      float* yp = Y + (m0 + 16 * mi + 8 * g) * (size_t)E_ + e;
#pragma unroll
      for (int r = 0; r < 8; ++r) yp[(size_t)r * E_] = acc[mi][ei][r] + bb;
    }
  }
}

extern "C" void kernel_launch(void* const* d_in, const int* in_sizes, int n_in,
                              void* d_out, int out_size, void* d_ws,
                              size_t ws_size, hipStream_t stream) {
  const float* values  = (const float*)d_in[0];
  const float* keysp   = (const float*)d_in[1];
  const float* queries = (const float*)d_in[2];
  const int*   mask    = (const int*)d_in[3];
  const float* W_out   = (const float*)d_in[4];
  const float* b_out   = (const float*)d_in[5];
  float* out = (float*)d_out;

  // workspace: [ X_bf16 : N*S*E ][ W_bf16 : E*E ]  (~168 MB)
  __bf16* Xbf = (__bf16*)d_ws;
  __bf16* Wbf = (__bf16*)((char*)d_ws + (size_t)N_ * S_ * E_ * sizeof(__bf16));

  cvt_w_kernel<<<(E_ * E_) / (256 * 4), 256, 0, stream>>>(W_out, Wbf);
  attn_softmax_kernel<<<(N_ * S_) / 4, 128, 0, stream>>>(values, keysp, queries,
                                                         mask, Xbf);
  out_proj_kernel<<<(N_ * S_ / BM_) * (E_ / BE_), 256, 0, stream>>>(Xbf, Wbf,
                                                                    b_out, out);
}
